// LocalAggregationLoss_36369783062989
// MI455X (gfx1250) — matrix operand order repaired and checked by reading, not compiled, not run on hardware
//
#include <hip/hip_runtime.h>
#include <stdint.h>

typedef __attribute__((ext_vector_type(16))) _Float16 v16h;
typedef __attribute__((ext_vector_type(8)))  float    v8f;

#define B_CODES 256
#define N_BANK  200000
#define D_DIM   128
// exp(x/T) = exp2(x * log2(e)/T)
#define EXP2_SCALE ((float)(1.4426950408889634 / 0.07))

// Workspace layout:
//   [0 .. 64KB)    : packed normalized codes (B-operand fragments, f16)
//                    layout [ct=16][kc=4][lane=32][16 halfs]
//   [64KB .. +1KB) : d1 accumulators (256 f32)
//   [65KB .. +1KB) : d2 accumulators (256 f32)

// ---------------------------------------------------------------------------
// Kernel 1: normalize code rows, pack into WMMA B-operand fragment order,
// zero the global density accumulators (ws is poisoned -> re-zero per launch).
//
// B-matrix (32x16 f16) per-lane layout used here (contiguous k, same as A):
//   lane L holds column n = L & 15; elements j=0..15 hold K = (L>>4)*16 + j
// ---------------------------------------------------------------------------
__global__ void __launch_bounds__(256)
la_prep(const float* __restrict__ codes,
        _Float16* __restrict__ cpack,
        float* __restrict__ d1, float* __restrict__ d2)
{
    int m = threadIdx.x;               // code row 0..255
    d1[m] = 0.0f;
    d2[m] = 0.0f;

    const float* row = codes + m * D_DIM;
    float ss = 0.0f;
#pragma unroll
    for (int i = 0; i < D_DIM; ++i) {
        float x = row[i];
        ss += x * x;
    }
    float inv = rsqrtf(ss);

    int mt = m >> 4;                   // code tile (n-tile of the GEMM)
    int l  = m & 15;                   // column index within the tile
#pragma unroll
    for (int kc = 0; kc < 4; ++kc) {   // K chunks of 32
        _Float16* p0 = cpack + (((mt * 4 + kc) * 32 + l)      << 4);
        _Float16* p1 = cpack + (((mt * 4 + kc) * 32 + l + 16) << 4);
#pragma unroll
        for (int j = 0; j < 16; ++j) {
            p0[j] = (_Float16)(row[kc * 32 + j]      * inv);
            p1[j] = (_Float16)(row[kc * 32 + 16 + j] * inv);
        }
    }
}

// ---------------------------------------------------------------------------
// Kernel 2: streaming GEMM over the bank (M = bank rows, N = codes), exp,
// masked reduction into per-code densities.
//
// A-matrix (16x32 f16) per-lane layout used here (contiguous k, same as B):
//   lane L holds row m = L & 15;
//   lane <16 : elems 0..15 = K{k0..k0+15}
//   lane >=16: elems 0..15 = K{k0+16..k0+31}
// C/D (16x16 f32): VGPR r, lane L -> M = r + 8*(L>>4), N = L & 15
//
// Dynamic LDS: [0..64KB) staged code fragments, [64KB..) s1/s2 accumulators.
// ---------------------------------------------------------------------------
__global__ void __launch_bounds__(256)
la_main(const float* __restrict__ bank,
        const unsigned char* __restrict__ mbg,
        const unsigned char* __restrict__ mit,
        const _Float16* __restrict__ cpack,
        float* __restrict__ d1, float* __restrict__ d2)
{
    extern __shared__ char smem[];
    _Float16* sb = (_Float16*)smem;                    // 64 KB fragment cache
    float* ls1 = (float*)(smem + 65536);               // 256 f32
    float* ls2 = ls1 + B_CODES;                        // 256 f32

    int t = threadIdx.x;

    // Stage the whole codes pack into LDS (L2 -> LDS, once per block).
    {
        const uint4* src = (const uint4*)cpack;
        uint4* dst = (uint4*)sb;
#pragma unroll
        for (int i = 0; i < 16; ++i)                   // 4096 uint4 / 256 thr
            dst[t + i * 256] = src[t + i * 256];
    }
    ls1[t] = 0.0f;
    ls2[t] = 0.0f;
    __syncthreads();

    int wave = t >> 5;
    int lane = t & 31;
    int sel  = lane >> 4;
    int li   = lane & 15;
    int mbase = blockIdx.x * 128 + wave * 16;          // first bank row of wave

    if (mbase < N_BANK) {                              // wave-uniform (N%16==0)
        // ---- A fragments: this wave's 16 bank rows, fp32 -> f16 ----
        const float* brow = bank + (size_t)(mbase + li) * D_DIM;
        v16h A0, A1, A2, A3;
        {
            int kofs = sel * 16;
#pragma unroll
            for (int j = 0; j < 8; ++j) {
                A0[j]     = (_Float16)brow[kofs + j];
                A0[8 + j] = (_Float16)brow[kofs + 8 + j];
                A1[j]     = (_Float16)brow[32 + kofs + j];
                A1[8 + j] = (_Float16)brow[32 + kofs + 8 + j];
                A2[j]     = (_Float16)brow[64 + kofs + j];
                A2[8 + j] = (_Float16)brow[64 + kofs + 8 + j];
                A3[j]     = (_Float16)brow[96 + kofs + j];
                A3[8 + j] = (_Float16)brow[96 + kofs + 8 + j];
            }
        }

        size_t mcol = (size_t)mbase + (size_t)(sel * 8);   // 8B aligned (N%8==0)
        const v16h* sfrag = (const v16h*)sb;

        for (int ct = 0; ct < 16; ++ct) {              // 16 code tiles (n)
            int nc = ct * 16 + li;                     // code index, this lane
            size_t moff = (size_t)nc * N_BANK + mcol;

            // Long-latency HBM mask gathers first.
            uint2 bg = *(const uint2*)(mbg + moff);
            uint2 it = *(const uint2*)(mit + moff);
            if (ct < 15) {                             // keep next tile near
                __builtin_prefetch(mbg + moff + (size_t)16 * N_BANK, 0, 3);
                __builtin_prefetch(mit + moff + (size_t)16 * N_BANK, 0, 3);
            }

            // B fragments from LDS (distinct regs -> one clause, one wait).
            const v16h* bb = sfrag + ct * 128 + lane;
            v16h Bf0 = bb[0];
            v16h Bf1 = bb[32];
            v16h Bf2 = bb[64];
            v16h Bf3 = bb[96];

            v8f c = {};
            c = __builtin_amdgcn_wmma_f32_16x16x32_f16(false, A0, false, Bf0,
                                                       (short)0, c, false, false);
            c = __builtin_amdgcn_wmma_f32_16x16x32_f16(false, A1, false, Bf1,
                                                       (short)0, c, false, false);
            c = __builtin_amdgcn_wmma_f32_16x16x32_f16(false, A2, false, Bf2,
                                                       (short)0, c, false, false);
            c = __builtin_amdgcn_wmma_f32_16x16x32_f16(false, A3, false, Bf3,
                                                       (short)0, c, false, false);

            // exp + masked accumulate (u32 bit tests, fma form).
            float a1 = 0.0f, a2 = 0.0f;
#pragma unroll
            for (int r = 0; r < 8; ++r) {
                uint32_t w1 = (r < 4) ? bg.x : bg.y;
                uint32_t w2 = (r < 4) ? it.x : it.y;
                float e = __builtin_amdgcn_exp2f(c[r] * EXP2_SCALE);
                a1 = fmaf((float)((w1 >> (8 * (r & 3))) & 1u), e, a1);
                a2 = fmaf((float)((w2 >> (8 * (r & 3))) & 1u), e, a2);
            }
            // combine lane halves (bank rows mbase+0..7 with +8..15)
            a1 += __shfl_xor(a1, 16);
            a2 += __shfl_xor(a2, 16);
            if (sel == 0) {
                atomicAdd(&ls1[nc], a1);               // ds_add_f32, no conflicts
                atomicAdd(&ls2[nc], a2);
            }
        }
    }

    __syncthreads();
    // one global atomic per code per block (~800K total over 512 addresses)
    atomicAdd(&d1[t], ls1[t]);
    atomicAdd(&d2[t], ls2[t]);
}

// ---------------------------------------------------------------------------
// Kernel 3: loss = mean( log d1 - log d2 )
// ---------------------------------------------------------------------------
__global__ void __launch_bounds__(256)
la_final(const float* __restrict__ d1, const float* __restrict__ d2,
         float* __restrict__ out)
{
    __shared__ float red[B_CODES];
    int t = threadIdx.x;
    red[t] = logf(d1[t]) - logf(d2[t]);
    __syncthreads();
#pragma unroll
    for (int s = 128; s > 0; s >>= 1) {
        if (t < s) red[t] += red[t + s];
        __syncthreads();
    }
    if (t == 0) out[0] = red[0] * (1.0f / (float)B_CODES);
}

// ---------------------------------------------------------------------------
extern "C" void kernel_launch(void* const* d_in, const int* in_sizes, int n_in,
                              void* d_out, int out_size, void* d_ws, size_t ws_size,
                              hipStream_t stream)
{
    const float* codes          = (const float*)d_in[0];          // [256,128] f32
    const float* bank           = (const float*)d_in[1];          // [200000,128] f32
    const unsigned char* mbg    = (const unsigned char*)d_in[2];  // [256,200000] bool
    const unsigned char* mit    = (const unsigned char*)d_in[3];  // [256,200000] bool

    _Float16* cpack = (_Float16*)d_ws;                            // 64 KB
    float* d1 = (float*)((char*)d_ws + 65536);                    // 256 f32
    float* d2 = d1 + B_CODES;                                     // 256 f32
    float* out = (float*)d_out;

    la_prep<<<1, 256, 0, stream>>>(codes, cpack, d1, d2);

    int nblocks = (N_BANK + 127) / 128;                           // 1563
    size_t shmem = 65536 + 2 * B_CODES * sizeof(float);           // 66 KB
    la_main<<<nblocks, 256, shmem, stream>>>(bank, mbg, mit, cpack, d1, d2);

    la_final<<<1, 256, 0, stream>>>(d1, d2, out);
}
